// VMFClassifier_45621142618871
// MI455X (gfx1250) — matrix operand order repaired and checked by reading, not compiled, not run on hardware
//
#include <hip/hip_runtime.h>
#include <stdint.h>

// ---------------- types for WMMA ----------------
typedef __attribute__((ext_vector_type(16))) __bf16 bf16x16;
typedef __attribute__((ext_vector_type(8)))  float  f32x8;

union BU { bf16x16 v; float4 f4[2]; };

#define DIMV   512
#define BATCH  512
#define KSMP   10
#define CCLS   8192
#define NCHUNK 8           // C chunks of 1024 columns, one wave each
#define ROWS   (BATCH*KSMP)

// ---------------- helpers ----------------
__device__ __forceinline__ uint32_t hashu(uint32_t x){
    x ^= x >> 16; x *= 0x7feb352dU; x ^= x >> 15; x *= 0x846ca68bU; x ^= x >> 16; return x;
}
__device__ __forceinline__ float u01(uint32_t h){
    return ((h >> 8) + 0.5f) * (1.0f/16777216.0f);
}
__device__ __forceinline__ float nrm01(uint32_t h1, uint32_t h2){
    float u1 = u01(h1), u2 = u01(h2);
    return sqrtf(-2.0f*__logf(u1)) * __cosf(6.28318530718f*u2);
}
__device__ __forceinline__ uint16_t f2bf(float f){
    uint32_t u = __float_as_uint(f);
    uint32_t r = (u + 0x7FFFu + ((u >> 16) & 1u)) >> 16;
    return (uint16_t)r;
}
__device__ __forceinline__ float vmf_logc_f(float k){
    const float nm14 = 127.75f, nm12 = 255.5f, np12 = 256.5f;
    float k2 = k*k;
    float sm = sqrtf(nm12*nm12 + k2);
    float sp = sqrtf(np12*np12 + k2);
    return nm14*(__logf(nm12 + sm) + __logf(nm12 + sp)) - 0.5f*(sm + sp);
}
__device__ __forceinline__ float blockReduceSum(float v, float* sred){
    for (int off = 16; off > 0; off >>= 1) v += __shfl_xor(v, off);
    int w = threadIdx.x >> 5, l = threadIdx.x & 31;
    __syncthreads();
    if (l == 0) sred[w] = v;
    __syncthreads();
    float r = 0.f;
    #pragma unroll
    for (int i = 0; i < 8; ++i) r += sred[i];
    return r;
}

// ---------------- preprocessing ----------------
__global__ void prep_x_kernel(const float* __restrict__ params,
                              float* __restrict__ muX, float* __restrict__ kapX){
    int wid  = (blockIdx.x*blockDim.x + threadIdx.x) >> 5;
    int lane = threadIdx.x & 31;
    if (wid >= BATCH) return;
    const float* row = params + (size_t)wid*(DIMV+2) + 1;
    float ss = 0.f;
    for (int i = lane; i < DIMV; i += 32){ float v = row[i]; ss += v*v; }
    for (int off = 16; off > 0; off >>= 1) ss += __shfl_xor(ss, off);
    float inv = rsqrtf(ss);
    for (int i = lane; i < DIMV; i += 32) muX[(size_t)wid*DIMV + i] = row[i]*inv;
    if (lane == 0) kapX[wid] = __expf(-params[(size_t)wid*(DIMV+2) + DIMV + 1]);
}

__global__ void prep_w_kernel(const float* __restrict__ weight,
                              const float* __restrict__ logScale,
                              uint16_t* __restrict__ muwBf,
                              float* __restrict__ normW,
                              float* __restrict__ aC, float* __restrict__ bC,
                              float* __restrict__ lcwC){
    int wid  = (blockIdx.x*blockDim.x + threadIdx.x) >> 5;
    int lane = threadIdx.x & 31;
    if (wid >= CCLS) return;
    const float* row = weight + (size_t)wid*(DIMV+2) + 1;
    float ss = 0.f;
    for (int i = lane; i < DIMV; i += 32){ float v = row[i]; ss += v*v; }
    for (int off = 16; off > 0; off >>= 1) ss += __shfl_xor(ss, off);
    float inv = rsqrtf(ss);
    for (int i = lane; i < DIMV; i += 32) muwBf[(size_t)wid*DIMV + i] = f2bf(row[i]*inv);
    if (lane == 0){
        float k2 = __expf(-weight[(size_t)wid*(DIMV+2) + DIMV + 1]);
        float k1 = __expf(logScale[0]);
        normW[wid] = ss*inv;               // sqrt(ss)
        aC[wid]   = k1*k1 + k2*k2;
        bC[wid]   = 2.0f*k1*k2;
        lcwC[wid] = vmf_logc_f(k2);
    }
}

// ---------------- vMF sampling (Wood's algorithm) ----------------
__global__ void sample_kernel(const float* __restrict__ muX,
                              const float* __restrict__ kapX,
                              uint16_t* __restrict__ smp){
    int b = blockIdx.x, t = threadIdx.x;
    __shared__ float sred[8];
    float mu0 = muX[(size_t)b*DIMV + t];
    float mu1 = muX[(size_t)b*DIMV + t + 256];
    float kap = kapX[b];
    float bb  = (-2.f*kap + sqrtf(4.f*kap*kap + 511.f*511.f)) / 511.f;
    float x0  = (1.f - bb)/(1.f + bb);
    float cc  = kap*x0 + 511.f*__logf(1.f - x0*x0);
    for (int k = 0; k < KSMP; ++k){
        uint32_t s = hashu((uint32_t)(b*KSMP + k)*0x9E3779B9u + 0x1234567u);
        // scalar rejection, computed redundantly (uniform across block)
        float wsel = 0.f, w0 = 0.f; bool found = false;
        for (int tt = 0; tt < 32; ++tt){
            uint32_t h0 = hashu(s + 3u*tt + 0u);
            uint32_t h1 = hashu(s + 3u*tt + 1u);
            uint32_t h2 = hashu(s + 3u*tt + 2u);
            float z = 0.5f + 0.0220970869f*nrm01(h0, h1);   // Beta(255.5,255.5) approx
            z = fminf(fmaxf(z, 1e-6f), 1.f - 1e-6f);
            float w = (1.f - (1.f + bb)*z) / (1.f - (1.f - bb)*z);
            float u = u01(h2);
            bool ok = (kap*w + 511.f*__logf(fmaxf(1.f - x0*w, 1e-37f)) - cc) >= __logf(u);
            if (tt == 0) w0 = w;
            if (ok && !found){ wsel = w; found = true; }
        }
        if (!found) wsel = w0;
        float sint = sqrtf(fmaxf(1.f - wsel*wsel, 0.f));
        // tangent direction
        float v0 = nrm01(hashu(s + 1000u + 4u*t), hashu(s + 1001u + 4u*t));
        float v1 = nrm01(hashu(s + 1002u + 4u*t), hashu(s + 1003u + 4u*t));
        float dotv = blockReduceSum(v0*mu0 + v1*mu1, sred);
        v0 -= dotv*mu0; v1 -= dotv*mu1;
        float nv  = blockReduceSum(v0*v0 + v1*v1, sred);
        float inv = rsqrtf(fmaxf(nv, 1e-30f));
        float s0 = wsel*mu0 + sint*v0*inv;
        float s1 = wsel*mu1 + sint*v1*inv;
        float nn = blockReduceSum(s0*s0 + s1*s1, sred);
        float in2 = rsqrtf(fmaxf(nn, 1e-30f));
        size_t base = ((size_t)b*KSMP + k)*DIMV;
        smp[base + t]       = f2bf(s0*in2);
        smp[base + t + 256] = f2bf(s1*in2);
    }
}

// ---------------- fused WMMA GEMM + logc + online logsumexp ----------------
__global__ __launch_bounds__(256, 1)
void gemm_lse_kernel(const uint16_t* __restrict__ smp,
                     const uint16_t* __restrict__ muw,
                     const float* __restrict__ aC, const float* __restrict__ bC,
                     const float* __restrict__ lcwC,
                     float* __restrict__ partial){
    const int lane  = threadIdx.x & 31;
    const int half  = lane >> 4;
    const int l     = lane & 15;
    const int chunk = threadIdx.x >> 5;       // 0..7 : 1024-column slice of C
    const int row0  = blockIdx.x * 16;
    const int m     = row0 + l;

    // Hold the full 16x512 bf16 A tile in registers (16 x v16bf).
    // A layout (ISA 7.12.2): lanes 0-15 hold K=[0..7,16..23], lanes 16-31 K=[8..15,24..31].
    bf16x16 A[16];
    #pragma unroll
    for (int kc = 0; kc < 16; ++kc){
        const char* p = (const char*)smp + ((size_t)m*DIMV + kc*32 + half*8)*2;
        BU u; u.f4[0] = *(const float4*)p; u.f4[1] = *(const float4*)(p + 32);
        A[kc] = u.v;
    }

    float mx[8], sv[8];
    #pragma unroll
    for (int r = 0; r < 8; ++r){ mx[r] = -3.0e38f; sv[r] = 0.f; }
    const float nm12 = 255.5f, np12 = 256.5f, nm14 = 127.75f;

    for (int ct = 0; ct < 64; ++ct){
        const int ccol = chunk*1024 + ct*16 + l;   // this lane's class column
        f32x8 acc = {};
        #pragma unroll
        for (int kc = 0; kc < 16; ++kc){
            // B layout: lanes 0-15 hold K=[0..7,16..23], lanes 16-31 K=[8..15,24..31] of column l
            const char* p = (const char*)muw + ((size_t)ccol*DIMV + kc*32 + half*8)*2;
            BU u; u.f4[0] = *(const float4*)p; u.f4[1] = *(const float4*)(p + 32);
            acc = __builtin_amdgcn_wmma_f32_16x16x32_bf16(
                    false, A[kc], false, u.v, (short)0, acc, false, false);
        }
        const float ac = aC[ccol], bc = bC[ccol], lcw = lcwC[ccol];
        #pragma unroll
        for (int r = 0; r < 8; ++r){
            float kk2 = fmaf(bc, acc[r], ac);                  // kk^2
            float sm_ = sqrtf(nm12*nm12 + kk2);
            float sp_ = sqrtf(np12*np12 + kk2);
            float val = lcw - nm14*(__logf(nm12 + sm_) + __logf(nm12 + sp_))
                            + 0.5f*(sm_ + sp_);                // logc_w - logc(kk)
            float nmx = fmaxf(mx[r], val);
            sv[r] = sv[r]*__expf(mx[r] - nmx) + __expf(val - nmx);
            mx[r] = nmx;
        }
    }
    // merge logsumexp state across the 16 lanes of each half (same rows, different cols)
    #pragma unroll
    for (int r = 0; r < 8; ++r){
        for (int off = 8; off > 0; off >>= 1){
            float omx = __shfl_xor(mx[r], off);
            float osv = __shfl_xor(sv[r], off);
            float nmx = fmaxf(mx[r], omx);
            sv[r] = sv[r]*__expf(mx[r]-nmx) + osv*__expf(omx-nmx);
            mx[r] = nmx;
        }
    }
    if (l == 0){
        #pragma unroll
        for (int r = 0; r < 8; ++r){
            int row = row0 + half*8 + r;
            size_t idx = ((size_t)row*NCHUNK + chunk)*2;
            partial[idx]   = mx[r];
            partial[idx+1] = sv[r];
        }
    }
}

// ---------------- output ----------------
__global__ void zero_kernel(float* __restrict__ out, size_t n){
    size_t i = (size_t)blockIdx.x*blockDim.x + threadIdx.x;
    size_t st = (size_t)gridDim.x*blockDim.x;
    for (; i < n; i += st) out[i] = 0.f;
}

__global__ void finalize_kernel(const float* __restrict__ partial,
                                const float* __restrict__ muX,
                                const float* __restrict__ weight,
                                const float* __restrict__ normW,
                                const float* __restrict__ logScale,
                                const int* __restrict__ labels,
                                float* __restrict__ out){
    int b    = (blockIdx.x*blockDim.x + threadIdx.x) >> 5;
    int lane = threadIdx.x & 31;
    if (b >= BATCH) return;
    int lab = labels[b];
    const float* wrow = weight + (size_t)lab*(DIMV+2) + 1;
    const float* mrow = muX + (size_t)b*DIMV;
    float d = 0.f;
    for (int i = lane; i < DIMV; i += 32) d += mrow[i]*wrow[i];
    for (int off = 16; off > 0; off >>= 1) d += __shfl_xor(d, off);
    if (lane == 0){
        float scale = __expf(logScale[0]);
        float neg_lognum = scale * d / normW[lab];
        float acc = 0.f;
        for (int k = 0; k < KSMP; ++k){
            size_t row = (size_t)b*KSMP + k;
            float m = -3.0e38f;
            for (int j = 0; j < NCHUNK; ++j) m = fmaxf(m, partial[(row*NCHUNK + j)*2]);
            float s = 0.f;
            for (int j = 0; j < NCHUNK; ++j)
                s += partial[(row*NCHUNK + j)*2 + 1] * __expf(partial[(row*NCHUNK + j)*2] - m);
            acc += m + __logf(s);      // neg_logden[b,k]  (logc(scale) terms cancel)
        }
        float loss = acc*(1.0f/KSMP) - neg_lognum;
        out[(size_t)b*CCLS + lab] = -loss;
    }
}

// ---------------- launcher ----------------
extern "C" void kernel_launch(void* const* d_in, const int* in_sizes, int n_in,
                              void* d_out, int out_size, void* d_ws, size_t ws_size,
                              hipStream_t stream){
    (void)in_sizes; (void)n_in; (void)ws_size;
    const float* params   = (const float*)d_in[0];
    const float* weight   = (const float*)d_in[1];
    const float* logScale = (const float*)d_in[2];
    const int*   labels   = (const int*)d_in[3];
    float* out = (float*)d_out;

    // workspace layout
    float*    muX    = (float*)d_ws;                       // 512*512
    float*    kapX   = muX + (size_t)BATCH*DIMV;           // 512
    float*    normW  = kapX + BATCH;                       // 8192
    float*    aC     = normW + CCLS;
    float*    bC     = aC + CCLS;
    float*    lcwC   = bC + CCLS;
    uint16_t* muwBf  = (uint16_t*)(lcwC + CCLS);           // 8192*512 bf16
    uint16_t* smpBf  = muwBf + (size_t)CCLS*DIMV;          // 5120*512 bf16
    float*    partial= (float*)(smpBf + (size_t)ROWS*DIMV);// 5120*8*2

    prep_x_kernel  <<<BATCH/8, 256, 0, stream>>>(params, muX, kapX);
    prep_w_kernel  <<<CCLS/8,  256, 0, stream>>>(weight, logScale, muwBf, normW, aC, bC, lcwC);
    sample_kernel  <<<BATCH,   256, 0, stream>>>(muX, kapX, smpBf);
    zero_kernel    <<<2048,    256, 0, stream>>>(out, (size_t)out_size);
    gemm_lse_kernel<<<ROWS/16, 256, 0, stream>>>(smpBf, muwBf, aC, bC, lcwC, partial);
    finalize_kernel<<<BATCH/8, 256, 0, stream>>>(partial, muX, weight, normW, logScale, labels, out);
}
